// RBFOperator_43069932044620
// MI455X (gfx1250) — matrix operand order repaired and checked by reading, not compiled, not run on hardware
//
#include <hip/hip_runtime.h>

// ---------------------------------------------------------------------------
// RBF operator, fused flash-attention style on gfx1250:
//   prep kernel : prototypes/proj (f32, reused 1024x) -> bf16 in d_ws, in the
//                 exact LDS tile layouts + precomputed ||p||^2.
//   main kernel : per 128-row x tile, stream P in blocks of 64:
//                 TDM (tensor_load_to_lds) double-buffered bf16 tile fetch,
//                 GEMM1 (x @ P^T) via v_wmma_f32_16x16x32_bf16,
//                 fused exp2(-dist/2) in C-registers,
//                 GEMM2 (phi @ proj) accumulating 16x256 f32 per wave.
// Roofline: 137 GFLOP vs 258 MB HBM -> matrix-unit bound; proto/proj stay in
// L2 (192MB); TDM halves L2 reads (bf16 once vs f32 per WG) and removes all
// per-block convert/transpose VALU.
// ---------------------------------------------------------------------------

typedef __attribute__((ext_vector_type(16))) __bf16 v16bf;
typedef __attribute__((ext_vector_type(8)))  __bf16 v8bf;
typedef __attribute__((ext_vector_type(8)))  float  v8f;
typedef __attribute__((ext_vector_type(4)))  float  v4f;
typedef __attribute__((ext_vector_type(4)))  unsigned int u32x4;
typedef __attribute__((ext_vector_type(8)))  unsigned int u32x8;

#define B_ROWS   131072
#define P_ROWS   1024
#define D_DIM    256
#define M_TILE   128              // 8 waves x 16 rows
#define PBLK     64               // prototypes per streamed block
#define NPB      (P_ROWS / PBLK)  // 16
#define X_STRIDE 264              // bf16 elems (528B rows -> bank-friendly)
#define P_STRIDE 264
#define PJ_STRIDE 72              // projT k-stride (64 + 8 pad)
#define PHI_STRIDE 72
#define NEG_HALF_LOG2E (-0.72134752044448170368f)  // exp(-d/2)=exp2(d*this)

// ---- workspace (d_ws) layout: bf16 tiles in LDS-ready layouts ----
#define WS_PROTO_ELEMS (P_ROWS * P_STRIDE)            // bf16 [1024][264]
#define WS_PROJT_ELEMS (NPB * D_DIM * PJ_STRIDE)      // bf16 [16][256][72]
#define WS_PROTO_OFF   0
#define WS_PROJT_OFF   (WS_PROTO_ELEMS * 2)           // 540672
#define WS_PSQ_OFF     (WS_PROJT_OFF + WS_PROJT_ELEMS * 2)  // 1130496

// ---- main-kernel LDS layout (bytes, all 16B aligned) ----
#define PROTO_TILE_B (PBLK * P_STRIDE * 2)            // 33792
#define PROJT_TILE_B (D_DIM * PJ_STRIDE * 2)          // 36864
#define BUF_STRIDE   (PROTO_TILE_B + PROJT_TILE_B)    // 70656
#define OFF_X    0
#define OFF_P0   (OFF_X + M_TILE * X_STRIDE * 2)      // 67584
#define OFF_PJ0  (OFF_P0 + PROTO_TILE_B)              // 101376
#define OFF_PHI  (OFF_P0 + 2 * BUF_STRIDE)            // 208896
#define OFF_XSQ  (OFF_PHI + 8 * 16 * PHI_STRIDE * 2)  // 227328
#define OFF_PART (OFF_XSQ + M_TILE * 4)               // 227840
#define LDS_BYTES (OFF_PART + 256 * 4)                // 228864

#define PROTO_TILE_DW (PROTO_TILE_B / 4)              // 8448  (fits tile_dim0 u16)
#define PROJT_TILE_DW (PROJT_TILE_B / 4)              // 9216

// ---------------------------------------------------------------------------
// TDM: 1-D contiguous tile copy global->LDS via Tensor Data Mover.
// D# group0/group1 bitfields per CDNA5 ISA §8.3/8.4. data_size=4B (code 2),
// tile = ndwords x 1. All operands wave-uniform -> SGPR tuples. EXEC ignored,
// so guard issuance by wave id. Tracked by TENSORcnt.
// ---------------------------------------------------------------------------
__device__ __forceinline__ void tdm_load_1d(unsigned ldsOff, const void* gsrc,
                                            unsigned ndwords) {
    unsigned long long ga = (unsigned long long)gsrc;
    u32x4 g0;
    g0[0] = 1u;                                   // count=1, user mode
    g0[1] = ldsOff;                               // lds_addr (bytes)
    g0[2] = (unsigned)ga;                         // global_addr[31:0]
    g0[3] = (unsigned)((ga >> 32) & 0x01FFFFFFu)  // global_addr[56:32]
            | (2u << 30);                         // type=2 ("image")
    u32x8 g1;
    g1[0] = 2u << 16;                             // data_size=4B; no mask/pad
    g1[1] = (ndwords & 0xFFFFu) << 16;            // tensor_dim0[15:0]
    g1[2] = ((ndwords >> 16) & 0xFFFFu)           // tensor_dim0[31:16]
            | (1u << 16);                         // tensor_dim1 = 1
    g1[3] = (ndwords & 0xFFFFu) << 16;            // tile_dim0 = ndwords
    g1[4] = 1u;                                   // tile_dim1=1, tile_dim2=0
    g1[5] = ndwords;                              // tensor_dim0_stride[31:0]
    g1[6] = 0u;
    g1[7] = 0u;
    asm volatile("tensor_load_to_lds %0, %1" : : "s"(g0), "s"(g1) : "memory");
}

__device__ __forceinline__ v16bf load_frag16(const __bf16* p0, const __bf16* p1) {
    v8bf a = *(const v8bf*)p0;
    v8bf b = *(const v8bf*)p1;
    v16bf r;
#pragma unroll
    for (int i = 0; i < 8; ++i) { r[i] = a[i]; r[i + 8] = b[i]; }
    return r;
}

// ---------------------------------------------------------------------------
// Prep: one block per prototype row p. Emits bf16 prototypes (stride 264),
// per-P-block transposed bf16 proj ([pb][n][k], k-stride 72), and ||p||^2.
// ---------------------------------------------------------------------------
__global__ __launch_bounds__(256)
void rbf_prep_kernel(const float* __restrict__ prot,
                     const float* __restrict__ proj,
                     __bf16* __restrict__ wsProto,
                     __bf16* __restrict__ wsProjT,
                     float* __restrict__ wsPsq) {
    __shared__ float red[256];
    const int p = blockIdx.x;    // 0..1023
    const int d = threadIdx.x;   // 0..255

    const float pv = prot[(size_t)p * D_DIM + d];
    wsProto[(size_t)p * P_STRIDE + d] = (__bf16)pv;
    red[d] = pv * pv;

    const float jv = proj[(size_t)p * D_DIM + d];
    wsProjT[(size_t)(p >> 6) * (D_DIM * PJ_STRIDE) + (size_t)d * PJ_STRIDE + (p & 63)] =
        (__bf16)jv;

    __syncthreads();
#pragma unroll
    for (int s = 128; s > 0; s >>= 1) {
        if (d < s) red[d] += red[d + s];
        __syncthreads();
    }
    if (d == 0) wsPsq[p] = red[0];
}

// ---------------------------------------------------------------------------
// Main fused kernel.
// ---------------------------------------------------------------------------
__global__ __launch_bounds__(256)
void rbf_fused_kernel(const float* __restrict__ x,
                      const __bf16* __restrict__ wsProto,
                      const __bf16* __restrict__ wsProjT,
                      const float* __restrict__ wsPsq,
                      float* __restrict__ out) {
    extern __shared__ char smem[];
    __bf16* ldsX   = (__bf16*)(smem + OFF_X);     // [128][264] bf16
    __bf16* ldsPhi = (__bf16*)(smem + OFF_PHI);   // [8][16][72] bf16 scratch
    float*  xsq    = (float*)(smem + OFF_XSQ);    // [128]
    float*  part   = (float*)(smem + OFF_PART);   // [256]

    const int tid   = threadIdx.x;
    const int mBase = blockIdx.x * M_TILE;
    const int wv    = tid >> 5;
    const int lane  = tid & 31;
    const int l     = lane & 15;
    const int lh    = lane >> 4;

    // ---- kick off TDM for block 0 into buffer 0 (overlaps with x staging)
    if (wv == 0) {
        tdm_load_1d(OFF_P0,  wsProto, PROTO_TILE_DW);
        tdm_load_1d(OFF_PJ0, wsProjT, PROJT_TILE_DW);
    }

    // ---- stage x tile as bf16 + per-row squared norms (2 threads / row) ----
    {
        const int r = tid >> 1, h = tid & 1;
        const float* src = x + (size_t)(mBase + r) * D_DIM + h * 128;
        __bf16* dst = ldsX + r * X_STRIDE + h * 128;
        float ss = 0.0f;
#pragma unroll
        for (int i = 0; i < 32; ++i) {
            v4f v = *(const v4f*)(src + i * 4);
            ss += v.x * v.x + v.y * v.y + v.z * v.z + v.w * v.w;
            dst[i * 4 + 0] = (__bf16)v.x;
            dst[i * 4 + 1] = (__bf16)v.y;
            dst[i * 4 + 2] = (__bf16)v.z;
            dst[i * 4 + 3] = (__bf16)v.w;
        }
        part[tid] = ss;
    }
    __syncthreads();
    if (tid < M_TILE) xsq[tid] = part[2 * tid] + part[2 * tid + 1];
    __syncthreads();

    // ---- hoist loop-invariant per-lane state -------------------------------
    float xs[8];
#pragma unroll
    for (int r = 0; r < 8; ++r) xs[r] = xsq[wv * 16 + 8 * lh + r];

    // GEMM1 A-fragments are invariant over the whole P loop: hoist (64 VGPRs).
    v16bf afrag[8];
#pragma unroll
    for (int ks = 0; ks < 8; ++ks) {
        const __bf16* arow = ldsX + (wv * 16 + l) * X_STRIDE + ks * 32;
        afrag[ks] = load_frag16(arow + lh * 8, arow + 16 + lh * 8);
    }

    v8f acc[16];
#pragma unroll
    for (int t = 0; t < 16; ++t) acc[t] = (v8f)0.0f;

    __bf16* phiW = ldsPhi + wv * 16 * PHI_STRIDE;

    // =================== stream over prototype blocks ======================
    for (int pb = 0; pb < NPB; ++pb) {
        const int pBase = pb * PBLK;

        // prefetch next block into the other buffer, then wait for this one
        if (wv == 0) {
            if (pb + 1 < NPB) {
                const unsigned nb = (unsigned)((pb + 1) & 1) * BUF_STRIDE;
                tdm_load_1d(OFF_P0 + nb,
                            wsProto + (size_t)(pb + 1) * PBLK * P_STRIDE,
                            PROTO_TILE_DW);
                tdm_load_1d(OFF_PJ0 + nb,
                            wsProjT + (size_t)(pb + 1) * D_DIM * PJ_STRIDE,
                            PROJT_TILE_DW);
                __builtin_amdgcn_s_wait_tensorcnt(2);  // this block's 2 ops done
            } else {
                __builtin_amdgcn_s_wait_tensorcnt(0);
            }
        }
        __syncthreads();   // tile pb visible to all waves

        const unsigned cb = (unsigned)(pb & 1) * BUF_STRIDE;
        const __bf16* ldsP  = (const __bf16*)(smem + OFF_P0 + cb);
        const __bf16* ldsPj = (const __bf16*)(smem + OFF_PJ0 + cb);

        // ---------------- GEMM1 + fused exp -> phi (bf16 in LDS) -----------
#pragma unroll
        for (int nt = 0; nt < 4; ++nt) {
            v8f c = (v8f)0.0f;
#pragma unroll
            for (int ks = 0; ks < 8; ++ks) {
                const __bf16* brow =
                    ldsP + (nt * 16 + l) * P_STRIDE + ks * 32 + lh * 8;
                v16bf b = load_frag16(brow, brow + 16);
                c = __builtin_amdgcn_wmma_f32_16x16x32_bf16(
                        false, afrag[ks], false, b, (short)0, c, false, false);
            }
            const float ps = wsPsq[pBase + nt * 16 + l];
#pragma unroll
            for (int r = 0; r < 8; ++r) {
                const float dist = xs[r] + ps - 2.0f * c[r];
                const float phi  = __builtin_amdgcn_exp2f(dist * NEG_HALF_LOG2E);
                phiW[(r + 8 * lh) * PHI_STRIDE + nt * 16 + l] = (__bf16)phi;
            }
        }

        // ---------------- GEMM2: acc += phi(16x64) @ proj(64x256) ----------
#pragma unroll
        for (int ks2 = 0; ks2 < 2; ++ks2) {
            const __bf16* arow = phiW + l * PHI_STRIDE + ks2 * 32;
            v16bf a = load_frag16(arow + lh * 8, arow + 16 + lh * 8);
#pragma unroll
            for (int nt2 = 0; nt2 < 16; ++nt2) {
                const __bf16* brow =
                    ldsPj + (nt2 * 16 + l) * PJ_STRIDE + ks2 * 32 + lh * 8;
                v16bf b = load_frag16(brow, brow + 16);
                acc[nt2] = __builtin_amdgcn_wmma_f32_16x16x32_bf16(
                               false, a, false, b, (short)0, acc[nt2], false, false);
            }
        }
        __syncthreads();   // all waves done with buffer (pb&1) before reuse
    }

    // ---- write output tile ------------------------------------------------
#pragma unroll
    for (int nt2 = 0; nt2 < 16; ++nt2) {
#pragma unroll
        for (int r = 0; r < 8; ++r) {
            out[(size_t)(mBase + wv * 16 + r + 8 * lh) * D_DIM + nt2 * 16 + l] =
                acc[nt2][r];
        }
    }
}

extern "C" void kernel_launch(void* const* d_in, const int* in_sizes, int n_in,
                              void* d_out, int out_size, void* d_ws, size_t ws_size,
                              hipStream_t stream) {
    const float* x    = (const float*)d_in[0];
    const float* prot = (const float*)d_in[1];
    const float* proj = (const float*)d_in[2];
    float* out = (float*)d_out;

    (void)in_sizes; (void)n_in; (void)out_size; (void)ws_size;

    __bf16* wsProto = (__bf16*)((char*)d_ws + WS_PROTO_OFF);
    __bf16* wsProjT = (__bf16*)((char*)d_ws + WS_PROJT_OFF);
    float*  wsPsq   = (float*)((char*)d_ws + WS_PSQ_OFF);   // ~1.13 MB total

    (void)hipFuncSetAttribute((const void*)rbf_fused_kernel,
                              hipFuncAttributeMaxDynamicSharedMemorySize,
                              LDS_BYTES);

    rbf_prep_kernel<<<dim3(P_ROWS), dim3(256), 0, stream>>>(
        prot, proj, wsProto, wsProjT, wsPsq);

    rbf_fused_kernel<<<dim3(B_ROWS / M_TILE), dim3(256), LDS_BYTES, stream>>>(
        x, wsProto, wsProjT, wsPsq, out);
}
